// Decoder_17076789969159
// MI455X (gfx1250) — compile-verified, matrix operand order checked by reading, not run on hardware
//
#include <hip/hip_runtime.h>

// ---------------------------------------------------------------------------
// MI455X / gfx1250 implementation.
// All heavy math runs through v_wmma_f32_16x16x32_bf16 (bf16 A/B, f32 acc).
// ---------------------------------------------------------------------------

typedef __attribute__((ext_vector_type(16))) __bf16 v16bf;
typedef __attribute__((ext_vector_type(8)))  float  v8f;

#define BN_EPS 1e-5f

__device__ __forceinline__ unsigned short f2bf(float f) {
    unsigned int u = __float_as_uint(f);
    u += 0x7FFFu + ((u >> 16) & 1u);       // round-to-nearest-even
    return (unsigned short)(u >> 16);
}

// ---------------------------------------------------------------------------
// Weight prep: w[Cout][Cin][9] f32  ->  Wt[9][Cout][cinPad] bf16 (zero-padded)
// ---------------------------------------------------------------------------
__global__ __launch_bounds__(256)
void prep_w_kernel(const float* __restrict__ w, unsigned short* __restrict__ wt,
                   int cout, int cin, int cinPad) {
    int i = blockIdx.x * 256 + threadIdx.x;
    int total = 9 * cout * cinPad;
    if (i >= total) return;
    int j   = i / (cout * cinPad);
    int rem = i - j * (cout * cinPad);
    int o   = rem / cinPad;
    int c   = rem - o * cinPad;
    float v = (c < cin) ? w[((size_t)o * cin + c) * 9 + j] : 0.f;
    wt[i] = f2bf(v);
}

// ---------------------------------------------------------------------------
// Stage 1: build h[b][n][c] (point-major, bf16, 320 padded channels)
//   c <  256 : linear interp of x (align_corners) 8192 -> N
//   256..303 : relu(bn1(conv1_w @ low_level_feat))
//   304..319 : zero pad
// ---------------------------------------------------------------------------
__global__ __launch_bounds__(256)
void stage1_kernel(const float* __restrict__ x, const float* __restrict__ llf,
                   const float* __restrict__ conv1_w,
                   const float* __restrict__ g1, const float* __restrict__ b1,
                   const float* __restrict__ m1, const float* __restrict__ v1,
                   unsigned short* __restrict__ hT, int B, int N, int L) {
    __shared__ float sllf[128];
    const int n = blockIdx.x, b = blockIdx.y, tid = threadIdx.x;
    if (tid < 128) sllf[tid] = llf[((size_t)b * 128 + tid) * N + n];
    __syncthreads();
    unsigned short* dst = hT + ((size_t)b * N + n) * 320;
    const float scale = (float)(L - 1) / (float)(N - 1);
    for (int c = tid; c < 320; c += 256) {
        float val = 0.f;
        if (c < 256) {
            float pos = n * scale;
            int i0 = (int)pos; if (i0 > L - 2) i0 = L - 2;
            float t = pos - (float)i0;
            const float* xr = x + ((size_t)b * 256 + c) * L;
            val = xr[i0] * (1.f - t) + xr[i0 + 1] * t;
        } else if (c < 304) {
            int cc = c - 256;
            const float* wr = conv1_w + cc * 128;
            float s = 0.f;
            for (int k = 0; k < 128; ++k) s += wr[k] * sllf[k];
            s = (s - m1[cc]) * (g1[cc] * rsqrtf(v1[cc] + BN_EPS)) + b1[cc];
            val = fmaxf(s, 0.f);
        }
        dst[c] = f2bf(val);
    }
}

// ---------------------------------------------------------------------------
// Multi-order weighted conv, one order per launch (sequential on stream).
//   hin : [B][N][cinPad] bf16 (point-major)
//   Wt  : [9][256][cinPad] bf16 (tap-major)
//   acc : [B][N][256] f32, original point order; beta=0 write, beta=1 add
// Workgroup = 256 thr = 8 waves. Tile = 256 couts x 64 permuted points.
// Wave w owns couts [32w, 32w+32) as 2 row-tiles x 4 col-tiles of WMMA.
// LDS holds 72 gathered rows (64 pts + 4-halo each side) + geometry.
// For each tap j: unweighted GEMM over cin, then scale result columns by
// Gaussian w(n,j) (one expf per lane, broadcast to the 8 acc rows).
// All fragment traffic is 16-byte aligned -> ds_load_b128 / global_load_b128.
// ---------------------------------------------------------------------------
__global__ __launch_bounds__(256)
void wconv_kernel(const unsigned short* __restrict__ hin,
                  const unsigned short* __restrict__ Wt,
                  const int* __restrict__ rot,
                  const float* __restrict__ coords,
                  const float* __restrict__ dist,
                  float* __restrict__ acc,
                  int ord, int B, int N, int O, int cinPad, int beta) {
    __shared__ __align__(16) unsigned short sh_act[72 * 320];
    __shared__ float sh_d[72], sh_c0[72], sh_c1[72], sh_c2[72];
    __shared__ int   sh_idx[72];

    const int tid   = threadIdx.x;
    const int b     = blockIdx.y;
    const int nBase = blockIdx.x * 64;
    const int* idx  = rot + ((size_t)ord * B + b) * N;

    // geometry + gather indices (halo gets poison distance -> weight == 0)
    for (int r = tid; r < 72; r += 256) {
        int p = nBase - 4 + r;
        if (p >= 0 && p < N) {
            int ip = idx[p];
            sh_idx[r] = ip;
            sh_d[r]   = dist[((size_t)b * O + ord) * N + p];
            sh_c0[r]  = coords[((size_t)b * 3 + 0) * N + ip];
            sh_c1[r]  = coords[((size_t)b * 3 + 1) * N + ip];
            sh_c2[r]  = coords[((size_t)b * 3 + 2) * N + ip];
        } else {
            sh_idx[r] = -1; sh_d[r] = 1e18f;
            sh_c0[r] = 0.f; sh_c1[r] = 0.f; sh_c2[r] = 0.f;
        }
    }
    __syncthreads();

    // gather permuted activation rows into LDS, 16 bytes per transaction
    const int qRow = cinPad >> 3;   // uint4 per row (40 or 32)
    const uint4* hin128 = (const uint4*)hin;
    uint4* sha128 = (uint4*)sh_act;
    const uint4 z4 = make_uint4(0u, 0u, 0u, 0u);
    for (int i = tid; i < 72 * qRow; i += 256) {
        int r = i / qRow, cq = i - r * qRow;
        int ip = sh_idx[r];
        sha128[r * qRow + cq] =
            (ip >= 0) ? hin128[((size_t)b * N + ip) * qRow + cq] : z4;
    }
    __syncthreads();

    const int wave = tid >> 5, lane = tid & 31;
    const int ln = lane & 15, hi = lane >> 4;
    const int coutBase = wave * 32;
    const int kcN = cinPad >> 5;

    v8f accP[2][4];
#pragma unroll
    for (int i = 0; i < 2; ++i)
#pragma unroll
        for (int ct = 0; ct < 4; ++ct)
#pragma unroll
            for (int r = 0; r < 8; ++r) accP[i][ct][r] = 0.f;

    for (int j = 0; j < 9; ++j) {
        v8f y[2][4];
#pragma unroll
        for (int i = 0; i < 2; ++i)
#pragma unroll
            for (int ct = 0; ct < 4; ++ct)
#pragma unroll
                for (int r = 0; r < 8; ++r) y[i][ct][r] = 0.f;

        for (int kc = 0; kc < kcN; ++kc) {
            const int kb = kc * 32 + hi * 8;  // bf16-element offset, per lane
            // A fragments: row m = coutBase(+16) + ln,
            // K halves [kb..kb+7] and [kb+16..kb+23] -> two aligned 16B chunks
            union { v16bf v; uint4 q[2]; } a0, a1;
            const uint4* ap0 = (const uint4*)
                (Wt + ((size_t)j * 256 + coutBase + ln) * cinPad + kb);
            const uint4* ap1 = (const uint4*)
                (Wt + ((size_t)j * 256 + coutBase + 16 + ln) * cinPad + kb);
            a0.q[0] = ap0[0]; a0.q[1] = ap0[2];
            a1.q[0] = ap1[0]; a1.q[1] = ap1[2];
#pragma unroll
            for (int ct = 0; ct < 4; ++ct) {
                // B fragment: column = local point ct*16+ln shifted by tap j
                const uint4* bp = (const uint4*)
                    (sh_act + (ct * 16 + ln + j) * cinPad + kb);
                union { v16bf v; uint4 q[2]; } bf;
                bf.q[0] = bp[0]; bf.q[1] = bp[2];
                y[0][ct] = __builtin_amdgcn_wmma_f32_16x16x32_bf16(
                    false, a0.v, false, bf.v, (short)0, y[0][ct], false, false);
                y[1][ct] = __builtin_amdgcn_wmma_f32_16x16x32_bf16(
                    false, a1.v, false, bf.v, (short)0, y[1][ct], false, false);
            }
        }
        // Gaussian geometry weight: per (output point, tap); column == lane
#pragma unroll
        for (int ct = 0; ct < 4; ++ct) {
            int oL = ct * 16 + ln;
            float dd = sh_d[oL + j]  - sh_d[oL + 4];
            float d0 = sh_c0[oL + j] - sh_c0[oL + 4];
            float d1 = sh_c1[oL + j] - sh_c1[oL + 4];
            float d2 = sh_c2[oL + j] - sh_c2[oL + 4];
            float w = __expf(-(dd * dd + d0 * d0 + d1 * d1 + d2 * d2));
#pragma unroll
            for (int r = 0; r < 8; ++r) {
                accP[0][ct][r] += w * y[0][ct][r];
                accP[1][ct][r] += w * y[1][ct][r];
            }
        }
    }

    // scatter back to original point order (idx is a permutation -> no races)
#pragma unroll
    for (int ct = 0; ct < 4; ++ct) {
        int orig = sh_idx[ct * 16 + ln + 4];
        float* drow = acc + ((size_t)b * N + orig) * 256;
#pragma unroll
        for (int rt = 0; rt < 2; ++rt) {
            float* d = drow + coutBase + rt * 16 + hi * 8;
            if (beta) {
#pragma unroll
                for (int r = 0; r < 8; ++r) d[r] += accP[rt][ct][r];
            } else {
#pragma unroll
                for (int r = 0; r < 8; ++r) d[r] = accP[rt][ct][r];
            }
        }
    }
}

// ---------------------------------------------------------------------------
// mean over orders + conv bias + BN + ReLU  ->  bf16 point-major [B][N][256]
// ---------------------------------------------------------------------------
__global__ __launch_bounds__(256)
void bnrelu_kernel(const float* __restrict__ acc, const float* __restrict__ bias,
                   const float* __restrict__ g, const float* __restrict__ bb,
                   const float* __restrict__ m, const float* __restrict__ v,
                   unsigned short* __restrict__ out, int B, int N) {
    int n = blockIdx.x, b = blockIdx.y, c = threadIdx.x;
    size_t i = ((size_t)b * N + n) * 256 + c;
    float val = acc[i] * (1.f / 3.f) + bias[c];
    val = (val - m[c]) * (g[c] * rsqrtf(v[c] + BN_EPS)) + bb[c];
    out[i] = f2bf(fmaxf(val, 0.f));
}

// ---------------------------------------------------------------------------
// final: mean + bias + BN2 + ReLU + 13x256 head  ->  out[B][13][N] f32
// ---------------------------------------------------------------------------
__global__ __launch_bounds__(256)
void final_kernel(const float* __restrict__ acc, const float* __restrict__ bias2,
                  const float* __restrict__ g, const float* __restrict__ bb,
                  const float* __restrict__ m, const float* __restrict__ v,
                  const float* __restrict__ ow, const float* __restrict__ ob,
                  float* __restrict__ out, int B, int N) {
    __shared__ float sv[256];
    int n = blockIdx.x, b = blockIdx.y, c = threadIdx.x;
    float val = acc[((size_t)b * N + n) * 256 + c] * (1.f / 3.f) + bias2[c];
    val = (val - m[c]) * (g[c] * rsqrtf(v[c] + BN_EPS)) + bb[c];
    sv[c] = fmaxf(val, 0.f);
    __syncthreads();
    if (c < 13) {
        float s = ob[c];
        const float* wr = ow + c * 256;
        for (int k = 0; k < 256; ++k) s += wr[k] * sv[k];
        out[((size_t)b * 13 + c) * N + n] = s;
    }
}

// ---------------------------------------------------------------------------
extern "C" void kernel_launch(void* const* d_in, const int* in_sizes, int n_in,
                              void* d_out, int out_size, void* d_ws, size_t ws_size,
                              hipStream_t stream) {
    const int B = 2, N = 32768, L = 8192, O = 3;
    const float* x       = (const float*)d_in[0];
    const float* llf     = (const float*)d_in[1];
    const float* coords  = (const float*)d_in[2];
    const int*   rot     = (const int*)d_in[3];
    const float* dist    = (const float*)d_in[4];
    const float* conv1_w = (const float*)d_in[5];
    const float* bn1g = (const float*)d_in[6],  *bn1b = (const float*)d_in[7];
    const float* bn1m = (const float*)d_in[8],  *bn1v = (const float*)d_in[9];
    const float* w1   = (const float*)d_in[10], *w1b  = (const float*)d_in[11];
    const float* c1g  = (const float*)d_in[12], *c1b  = (const float*)d_in[13];
    const float* c1m  = (const float*)d_in[14], *c1v  = (const float*)d_in[15];
    const float* w2   = (const float*)d_in[16], *w2b  = (const float*)d_in[17];
    const float* c2g  = (const float*)d_in[18], *c2b  = (const float*)d_in[19];
    const float* c2m  = (const float*)d_in[20], *c2v  = (const float*)d_in[21];
    const float* ow   = (const float*)d_in[22], *ob   = (const float*)d_in[23];
    (void)in_sizes; (void)n_in; (void)out_size; (void)ws_size;

    char* ws = (char*)d_ws;
    size_t off = 0;
    auto take = [&](size_t bytes) -> char* {
        char* p = ws + off;
        off = (off + bytes + 255) & ~(size_t)255;
        return p;
    };
    unsigned short* hT  = (unsigned short*)take((size_t)B * N * 320 * 2); // 40 MB
    float*          acc = (float*)         take((size_t)B * N * 256 * 4); // 64 MB
    unsigned short* h2  = (unsigned short*)take((size_t)B * N * 256 * 2); // 32 MB
    unsigned short* Wt1 = (unsigned short*)take((size_t)9 * 256 * 320 * 2);
    unsigned short* Wt2 = (unsigned short*)take((size_t)9 * 256 * 256 * 2);

    prep_w_kernel<<<(9 * 256 * 320 + 255) / 256, 256, 0, stream>>>(w1, Wt1, 256, 304, 320);
    prep_w_kernel<<<(9 * 256 * 256 + 255) / 256, 256, 0, stream>>>(w2, Wt2, 256, 256, 256);

    stage1_kernel<<<dim3(N, B), 256, 0, stream>>>(
        x, llf, conv1_w, bn1g, bn1b, bn1m, bn1v, hT, B, N, L);

    for (int o = 0; o < O; ++o)
        wconv_kernel<<<dim3(N / 64, B), 256, 0, stream>>>(
            hT, Wt1, rot, coords, dist, acc, o, B, N, O, 320, (o > 0) ? 1 : 0);

    bnrelu_kernel<<<dim3(N, B), 256, 0, stream>>>(acc, w1b, c1g, c1b, c1m, c1v, h2, B, N);

    for (int o = 0; o < O; ++o)
        wconv_kernel<<<dim3(N / 64, B), 256, 0, stream>>>(
            h2, Wt2, rot, coords, dist, acc, o, B, N, O, 256, (o > 0) ? 1 : 0);

    final_kernel<<<dim3(N, B), 256, 0, stream>>>(
        acc, w2b, c2g, c2b, c2m, c2v, ow, ob, (float*)d_out, B, N);
}
